// DeformableConvolutional2D_50388556316775
// MI455X (gfx1250) — matrix operand order repaired and checked by reading, not compiled, not run on hardware
//
#include <hip/hip_runtime.h>
#include <hip/hip_bf16.h>

typedef __attribute__((ext_vector_type(16))) _Float16 v16h;
typedef __attribute__((ext_vector_type(8)))  _Float16 v8h;
typedef __attribute__((ext_vector_type(4)))  _Float16 v4h;
typedef __attribute__((ext_vector_type(8)))  float    v8f;

#define BB 8
#define HH 128
#define WW 128
#define CC 64
#define FF 128
#define KK 576          // 9 * 64 contraction length
#define ASTRIDE 584     // padded LDS row stride (halves): bank-conflict-free
#define OTILE 32        // pixels per offset-GEMM workgroup

// ---------------------------------------------------------------------------
// Kernel 0: weight prep.
//   conv_W (3,3,C,F) f32 -> Wt  (F ,KK) f16   (B^T for main WMMA)
//   offset_W (3,3,C,9) f32 -> oWt (16,KK) f16 (B^T, rows 9..15 zero-padded)
// ---------------------------------------------------------------------------
__global__ __launch_bounds__(256) void wt_kernel(const float* __restrict__ cW,
                                                 const float* __restrict__ oW,
                                                 _Float16* __restrict__ wt,
                                                 _Float16* __restrict__ owt) {
    int t = blockIdx.x * 256 + threadIdx.x;
    if (t < KK * FF) {
        int f = t / KK;
        int k = t - f * KK;
        wt[t] = (_Float16)cW[(size_t)k * FF + f];
    }
    int t2 = t - KK * FF;
    if (t2 >= 0 && t2 < 16 * KK) {
        int o = t2 / KK;
        int k = t2 - o * KK;
        owt[t2] = (o < 9) ? (_Float16)oW[(size_t)k * 9 + o] : (_Float16)0.0f;
    }
}

// ---------------------------------------------------------------------------
// Kernel 1: offset conv as WMMA GEMM.  M=32 pixels/WG, K=576, N=16 (9 valid).
// Zero-padded ('SAME') im2col built in LDS; 2 wave32s, one 16-pixel tile each.
// ---------------------------------------------------------------------------
__global__ __launch_bounds__(64) void offset_gemm(const float* __restrict__ x,
                                                  const _Float16* __restrict__ owt,
                                                  const float* __restrict__ ob,
                                                  float* __restrict__ offs) {
    __shared__ __align__(16) _Float16 Atile[OTILE * ASTRIDE];   // ~36.5 KB

    int pixbase = blockIdx.x * OTILE;
    int b     = pixbase >> 14;
    int i     = (pixbase >> 7) & 127;
    int jbase = pixbase & 127;          // 128 % OTILE == 0 -> single row

    int tid = threadIdx.x;
    const float* xb = x + (size_t)b * HH * WW * CC;

    // ---- build zero-padded im2col A tile: 32 pixels x 576 K
#pragma unroll
    for (int tap = 0; tap < 9; ++tap) {
        int p = tap / 3, q = tap - 3 * (tap / 3);
        int ii = i + p - 1;
        bool rowok = (ii >= 0) && (ii < HH);
        for (int t = tid; t < OTILE * 16; t += 64) {
            int pix = t >> 4;
            int cg  = t & 15;
            int jj  = jbase + pix + q - 1;
            v4h r = {};
            if (rowok && jj >= 0 && jj < WW) {
                float4 s = *(const float4*)(xb + ((size_t)ii * WW + jj) * CC + cg * 4);
                r[0] = (_Float16)s.x; r[1] = (_Float16)s.y;
                r[2] = (_Float16)s.z; r[3] = (_Float16)s.w;
            }
            *(v4h*)&Atile[pix * ASTRIDE + tap * CC + cg * 4] = r;
        }
    }
    __syncthreads();

    // ---- WMMA: wave wv handles pixels [16wv, 16wv+16)
    int lane = tid & 31;
    int wv   = tid >> 5;               // 0..1
    int n    = lane & 15;
    int hiH  = lane >> 4;

    const _Float16* Arow = &Atile[(wv * 16 + n) * ASTRIDE + hiH * 8];
    const _Float16* Brow = owt + (size_t)n * KK + hiH * 8;

    v8f acc = {};
#pragma unroll
    for (int kc = 0; kc < KK; kc += 32) {
        v8h a0 = *(const v8h*)(Arow + kc);
        v8h a1 = *(const v8h*)(Arow + kc + 16);
        v16h av, bv;
#pragma unroll
        for (int e = 0; e < 8; ++e) { av[e] = a0[e]; av[e + 8] = a1[e]; }
        v8h b0 = *(const v8h*)(Brow + kc);
        v8h b1 = *(const v8h*)(Brow + kc + 16);
#pragma unroll
        for (int e = 0; e < 8; ++e) { bv[e] = b0[e]; bv[e + 8] = b1[e]; }
        acc = __builtin_amdgcn_wmma_f32_16x16x32_f16(
                  false, av, false, bv, (short)0, acc, false, false);
    }

    if (n < 9) {
        float bv_ = ob[n];
#pragma unroll
        for (int r = 0; r < 8; ++r) {
            int M = r + hiH * 8;
            offs[((size_t)(pixbase + wv * 16 + M)) * 9 + n] = acc[r] + bv_;
        }
    }
}

// ---------------------------------------------------------------------------
// Kernel 2: deformable sampling (bilinear im2col into LDS, f16) + WMMA GEMM.
// Workgroup = 16 output pixels x all 128 features; 8 wave32s, one N-tile each.
// ---------------------------------------------------------------------------
__global__ __launch_bounds__(256) void deform_gemm(const float* __restrict__ x,
                                                   const float* __restrict__ offs,
                                                   const _Float16* __restrict__ wt,
                                                   const float* __restrict__ bias,
                                                   float* __restrict__ out) {
    __shared__ __align__(16) _Float16 Atile[16 * ASTRIDE];   // ~18.7 KB

    int tile    = blockIdx.x;          // 0..8191
    int pixbase = tile * 16;
    int b     = pixbase >> 14;
    int i     = (pixbase >> 7) & 127;
    int jbase = pixbase & 127;

    int tid = threadIdx.x;

    // ---- build sampled A tile: 16 pixels x 576 K (tap-major, channel inner)
    {
        int pix = tid >> 4;            // 0..15 : pixel within tile
        int cg  = tid & 15;            // channel group -> c = 4*cg
        int j   = jbase + pix;
        const float* xb   = x + (size_t)b * HH * WW * CC;
        const float* offp = offs + ((size_t)(pixbase + pix)) * 9;
#pragma unroll
        for (int tap = 0; tap < 9; ++tap) {
            int p = tap / 3, q = tap - 3 * (tap / 3);
            int row = min(max(i + p - 1, 0), HH - 1);          // clamped rows
            float off = offp[tap];
            float xf  = (float)(j + q - 1) + off;
            float x0f = floorf(xf);
            float w1  = xf - x0f;
            int x0i = min(max((int)x0f, 0), WW - 1);
            int x1i = min(x0i + 1, WW - 1);
            const float4 s0 = *(const float4*)(xb + ((size_t)row * WW + x0i) * CC + cg * 4);
            const float4 s1 = *(const float4*)(xb + ((size_t)row * WW + x1i) * CC + cg * 4);
            v4h r;
            r[0] = (_Float16)fmaf(w1, s1.x - s0.x, s0.x);
            r[1] = (_Float16)fmaf(w1, s1.y - s0.y, s0.y);
            r[2] = (_Float16)fmaf(w1, s1.z - s0.z, s0.z);
            r[3] = (_Float16)fmaf(w1, s1.w - s0.w, s0.w);
            *(v4h*)&Atile[pix * ASTRIDE + tap * CC + cg * 4] = r;
        }
    }
    __syncthreads();

    // ---- WMMA: each wave does 16x16 output, contracting K=576 in 18 chunks
    int lane = tid & 31;
    int wv   = tid >> 5;               // 0..7 -> feature tile
    int n    = lane & 15;
    int hiH  = (lane >> 4);

    const _Float16* Arow = &Atile[n * ASTRIDE + hiH * 8];
    const _Float16* Brow = wt + ((size_t)(wv * 16 + n)) * KK + hiH * 8;

    v8f acc = {};
#pragma unroll
    for (int kc = 0; kc < KK; kc += 32) {
        v8h a0 = *(const v8h*)(Arow + kc);        // K = kc+klo   .. +7
        v8h a1 = *(const v8h*)(Arow + kc + 16);   // K = kc+16+klo.. +7
        v16h av, bv;
#pragma unroll
        for (int e = 0; e < 8; ++e) { av[e] = a0[e]; av[e + 8] = a1[e]; }
        v8h b0 = *(const v8h*)(Brow + kc);        // K = kc+klo   .. +7
        v8h b1 = *(const v8h*)(Brow + kc + 16);   // K = kc+16+klo.. +7
#pragma unroll
        for (int e = 0; e < 8; ++e) { bv[e] = b0[e]; bv[e + 8] = b1[e]; }
        acc = __builtin_amdgcn_wmma_f32_16x16x32_f16(
                  false, av, false, bv, (short)0, acc, false, false);
    }

    // ---- epilogue: +bias, scatter per C/D layout (VGPR r -> M = r + 8*hiH)
    float bv_ = bias[wv * 16 + n];
    size_t obase = (((size_t)b * HH + i) * WW + jbase) * FF + wv * 16 + n;
#pragma unroll
    for (int r = 0; r < 8; ++r) {
        int M = r + hiH * 8;
        out[obase + (size_t)M * FF] = acc[r] + bv_;
    }
}

// ---------------------------------------------------------------------------
extern "C" void kernel_launch(void* const* d_in, const int* in_sizes, int n_in,
                              void* d_out, int out_size, void* d_ws, size_t ws_size,
                              hipStream_t stream) {
    const float* x  = (const float*)d_in[0];   // (8,128,128,64)
    const float* oW = (const float*)d_in[1];   // (3,3,64,9)
    const float* ob = (const float*)d_in[2];   // (9)
    const float* cW = (const float*)d_in[3];   // (3,3,64,128)
    const float* cb = (const float*)d_in[4];   // (128)
    float* out = (float*)d_out;                // (8,128,128,128)

    const size_t nPix = (size_t)BB * HH * WW;            // 131072
    float*    offs = (float*)d_ws;                       // nPix*9 f32 = 4.5 MB
    _Float16* wt   = (_Float16*)((char*)d_ws + nPix * 9 * sizeof(float));
    _Float16* owt  = wt + (size_t)FF * KK;               // 16*KK f16

    wt_kernel  <<<(KK * FF + 16 * KK + 255) / 256, 256, 0, stream>>>(cW, oW, wt, owt);
    offset_gemm<<<(int)(nPix / OTILE), 64,  0, stream>>>(x, owt, ob, offs);
    deform_gemm<<<(int)(nPix / 16),    256, 0, stream>>>(x, offs, wt, cb, out);
}
